// InvariantPoseAttentionFn_7988639171221
// MI455X (gfx1250) — matrix operand order repaired and checked by reading, not compiled, not run on hardware
//
#include <hip/hip_runtime.h>
#include <hip/hip_bf16.h>

typedef _Float16 f16;
typedef __attribute__((ext_vector_type(16))) _Float16 v16h;
typedef __attribute__((ext_vector_type(8)))  _Float16 v8h;
typedef __attribute__((ext_vector_type(8)))  float    v8f;
typedef __attribute__((__vector_size__(16))) int      i32x4;

#define Bb 2
#define Hh 8
#define NQ 2048
#define NK 4096
#define CC 64
#define NCH (NK / 64)
#define SCALE 0.125f
#define OUT_ELEMS ((size_t)Bb * Hh * NQ * CC)

#if __has_builtin(__builtin_amdgcn_global_load_async_to_lds_b128)
#define IPA_ASYNC 1
#else
#define IPA_ASYNC 0
#endif

__device__ __forceinline__ v8f wmma16(v16h a, v16h b, v8f c) {
  return __builtin_amdgcn_wmma_f32_16x16x32_f16(false, a, false, b, (short)0, c,
                                                false, false);
}

__device__ __forceinline__ v16h ld_pair(const f16* p0, const f16* p1) {
  v8h a = *(const v8h*)p0;
  v8h b = *(const v8h*)p1;
  v16h r;
#pragma unroll
  for (int i = 0; i < 8; ++i) { r[i] = a[i]; r[i + 8] = b[i]; }
  return r;
}

__device__ __forceinline__ v16h ld_c16(const f16* p) {  // 16 contiguous halves
  return ld_pair(p, p + 8);
}

// 16-byte global -> LDS copy: async path if available, else via VGPRs.
__device__ __forceinline__ void cp16(f16* lds_dst, const f16* g_src) {
#if IPA_ASYNC
  __builtin_amdgcn_global_load_async_to_lds_b128(
      (__attribute__((address_space(1))) i32x4*)g_src,
      (__attribute__((address_space(3))) i32x4*)lds_dst, 0, 0);
#else
  *(uint4*)lds_dst = *(const uint4*)g_src;
#endif
}

__device__ __forceinline__ void async_wait0() {
#if IPA_ASYNC
#if __has_builtin(__builtin_amdgcn_s_wait_asynccnt)
  __builtin_amdgcn_s_wait_asynccnt(0);
#else
  asm volatile("s_wait_asynccnt 0" ::: "memory");
#endif
#endif
}

// ---- Prep: kx = Z * k4 (f16), vT = v transposed [bh][C][Nk] (f16) ----
__global__ __launch_bounds__(256) void ipa_prep_kv(
    const float* __restrict__ k, const float* __restrict__ v,
    const float* __restrict__ zR, const float* __restrict__ zT,
    f16* __restrict__ kx, f16* __restrict__ vT) {
  int idx = blockIdx.x * 256 + threadIdx.x;
  {  // kx: [bh:4][n:12][g:4] -> coalesced k reads and kx writes
    int g = idx & 15;
    int n = (idx >> 4) & (NK - 1);
    int bh = idx >> 16;
    int b = bh >> 3;
    const float* kp = k + ((size_t)bh * NK + n) * CC + g * 4;
    float u0 = kp[0], u1 = kp[1], u2 = kp[2], u3 = kp[3];
    const float* R = zR + ((size_t)b * NK + n) * 9;
    const float* t = zT + ((size_t)b * NK + n) * 3;
    float o0 = R[0] * u0 + R[1] * u1 + R[2] * u2 + t[0] * u3;
    float o1 = R[3] * u0 + R[4] * u1 + R[5] * u2 + t[1] * u3;
    float o2 = R[6] * u0 + R[7] * u1 + R[8] * u2 + t[2] * u3;
    f16* ko = kx + ((size_t)bh * NK + n) * CC + g * 4;
    ko[0] = (f16)o0; ko[1] = (f16)o1; ko[2] = (f16)o2; ko[3] = (f16)u3;
  }
  {  // vT: [bh:4][g:4][n:12] -> coalesced vT writes (64B per wave per channel)
    int n = idx & (NK - 1);
    int g = (idx >> 12) & 15;
    int bh = idx >> 16;
    const float* vp = v + ((size_t)bh * NK + n) * CC + g * 4;
    f16* vo = vT + ((size_t)bh * CC + g * 4) * NK + n;
    vo[0]      = (f16)vp[0];
    vo[NK]     = (f16)vp[1];
    vo[2 * NK] = (f16)vp[2];
    vo[3 * NK] = (f16)vp[3];
  }
}

// ---- Prep: qx = (Xinv)^T * q4 (f16) ----
__global__ __launch_bounds__(256) void ipa_prep_q(
    const float* __restrict__ q, const float* __restrict__ xR,
    const float* __restrict__ xT, f16* __restrict__ qx) {
  int idx = blockIdx.x * 256 + threadIdx.x;  // [bh:4][n:11][g:4]
  int g = idx & 15;
  int n = (idx >> 4) & (NQ - 1);
  int bh = idx >> 15;
  int b = bh >> 3;
  const float* qp = q + ((size_t)bh * NQ + n) * CC + g * 4;
  float u0 = qp[0], u1 = qp[1], u2 = qp[2], u3 = qp[3];
  const float* R = xR + ((size_t)b * NQ + n) * 9;
  const float* t = xT + ((size_t)b * NQ + n) * 3;
  // qx[i<3] = R u ;  qx[3] = u3 - (R^T t) . u[0:3]
  float o0 = R[0] * u0 + R[1] * u1 + R[2] * u2;
  float o1 = R[3] * u0 + R[4] * u1 + R[5] * u2;
  float o2 = R[6] * u0 + R[7] * u1 + R[8] * u2;
  float w0 = R[0] * t[0] + R[3] * t[1] + R[6] * t[2];
  float w1 = R[1] * t[0] + R[4] * t[1] + R[7] * t[2];
  float w2 = R[2] * t[0] + R[5] * t[1] + R[8] * t[2];
  float o3 = u3 - (w0 * u0 + w1 * u1 + w2 * u2);
  f16* qo = qx + ((size_t)bh * NQ + n) * CC + g * 4;
  qo[0] = (f16)o0; qo[1] = (f16)o1; qo[2] = (f16)o2; qo[3] = (f16)o3;
}

// ---- Attention: per block = (bh, 128 q rows); per wave = 16 q rows ----
__global__ __launch_bounds__(256) void ipa_attn(
    const f16* __restrict__ qx, const f16* __restrict__ kx,
    const f16* __restrict__ vT, float* __restrict__ out,
    float* __restrict__ attn) {
  __shared__ f16 kxs[2][64 * 72];  // double-buffered [64 keys][64 ch + pad]
  __shared__ f16 vTs[2][64 * 72];  // double-buffered [64 ch][64 keys + pad]
  __shared__ f16 ps[8][16 * 40];   // per-wave P tile [16 rows][32 keys + pad]

  const int tid  = threadIdx.x;
  const int wave = tid >> 5;
  const int lane = tid & 31;
  const int l16  = lane & 15;
  const int hi   = lane >> 4;  // 0: K 0..15 / rows 0..7
  const int aOff = hi ? 8 : 0;

  const int bh = blockIdx.x >> 4;
  const int qb = blockIdx.x & 15;
  const int rowBase = qb * 128 + wave * 16;

  const f16* kxg = kx + (size_t)bh * NK * CC;
  const f16* vTg = vT + (size_t)bh * CC * NK;

  // A fragments of qx (two 32-wide K chunks along C)
  const f16* qrow = qx + ((size_t)bh * NQ + rowBase + l16) * CC;
  const v16h A0 = ld_c16(qrow + hi * 16);
  const v16h A1 = ld_c16(qrow + 32 + hi * 16);

  auto stage_kx = [&](int buf, int c0) {
#pragma unroll
    for (int i = tid; i < 512; i += 256) {
      int key = i >> 3, seg = i & 7;
      cp16(&kxs[buf][key * 72 + seg * 8],
           &kxg[(size_t)(c0 + key) * CC + seg * 8]);
    }
  };
  auto stage_v = [&](int buf, int c0) {
#pragma unroll
    for (int i = tid; i < 512; i += 256) {
      int ch = i >> 3, seg = i & 7;
      cp16(&vTs[buf][ch * 72 + seg * 8],
           &vTg[(size_t)ch * NK + c0 + seg * 8]);
    }
  };

  float m[8], lsum[8];
#pragma unroll
  for (int v = 0; v < 8; ++v) { m[v] = -1e30f; lsum[v] = 0.0f; }

  // -------- Pass 1: online (max, sumexp) --------
  stage_kx(0, 0);
  async_wait0();
  __syncthreads();
  for (int ci = 0; ci < NCH; ++ci) {
    const int cur = ci & 1;
    if (ci + 1 < NCH) stage_kx(1 - cur, (ci + 1) * 64);  // overlaps compute
    const f16* kb = &kxs[cur][0];
#pragma unroll
    for (int kt = 0; kt < 4; ++kt) {
      const f16* bp = &kb[(kt * 16 + l16) * 72 + hi * 16];
      v8f S = {};
      S = wmma16(A0, ld_c16(bp),      S);
      S = wmma16(A1, ld_c16(bp + 32), S);
#pragma unroll
      for (int v = 0; v < 8; ++v) {
        float s  = S[v] * SCALE;
        float mn = fmaxf(m[v], s);
        lsum[v]  = lsum[v] * __expf(m[v] - mn) + __expf(s - mn);
        m[v]     = mn;
      }
    }
    async_wait0();
    __syncthreads();
  }
  // combine across the 16 lanes sharing each row
  for (int d = 1; d < 16; d <<= 1) {
#pragma unroll
    for (int v = 0; v < 8; ++v) {
      float om = __shfl_xor(m[v], d, 32);
      float ol = __shfl_xor(lsum[v], d, 32);
      float mn = fmaxf(m[v], om);
      lsum[v]  = lsum[v] * __expf(m[v] - mn) + ol * __expf(om - mn);
      m[v]     = mn;
    }
  }
  float rinv[8];
#pragma unroll
  for (int v = 0; v < 8; ++v) rinv[v] = 1.0f / lsum[v];

  // -------- Pass 2: recompute S, write attn (NT), accumulate P*V --------
  v8f acc[4] = {v8f{}, v8f{}, v8f{}, v8f{}};
  f16* pw = &ps[wave][0];
  stage_kx(0, 0);
  stage_v(0, 0);
  async_wait0();
  __syncthreads();
  for (int ci = 0; ci < NCH; ++ci) {
    const int cur = ci & 1;
    const int c0 = ci * 64;
    if (ci + 1 < NCH) {
      stage_kx(1 - cur, c0 + 64);
      stage_v(1 - cur, c0 + 64);
    }
    const f16* kb = &kxs[cur][0];
    const f16* vb0 = &vTs[cur][0];
#pragma unroll
    for (int g = 0; g < 2; ++g) {  // two 32-key groups
#pragma unroll
      for (int kt = 0; kt < 2; ++kt) {
        int ktile = g * 2 + kt;
        const f16* bp = &kb[(ktile * 16 + l16) * 72 + hi * 16];
        v8f S = {};
        S = wmma16(A0, ld_c16(bp),      S);
        S = wmma16(A1, ld_c16(bp + 32), S);
        int col = c0 + ktile * 16 + l16;
#pragma unroll
        for (int v = 0; v < 8; ++v) {
          int r   = hi * 8 + v;
          float p = __expf(S[v] * SCALE - m[v]) * rinv[v];
          __builtin_nontemporal_store(
              p, &attn[((size_t)bh * NQ + rowBase + r) * NK + col]);
          pw[r * 40 + kt * 16 + l16] = (f16)p;
        }
      }
      asm volatile("s_wait_dscnt 0" ::: "memory");  // P stores visible in-wave
      const f16* pr = &pw[l16 * 40];
      v16h PA = ld_c16(pr + hi * 16);
#pragma unroll
      for (int ct = 0; ct < 4; ++ct) {
        const f16* vb = &vb0[(ct * 16 + l16) * 72 + g * 32 + hi * 16];
        acc[ct] = wmma16(PA, ld_c16(vb), acc[ct]);
      }
    }
    async_wait0();
    __syncthreads();
  }

#pragma unroll
  for (int ct = 0; ct < 4; ++ct)
#pragma unroll
    for (int v = 0; v < 8; ++v)
      __builtin_nontemporal_store(
          acc[ct][v],
          &out[((size_t)bh * NQ + rowBase + hi * 8 + v) * CC + ct * 16 + l16]);
}

extern "C" void kernel_launch(void* const* d_in, const int* in_sizes, int n_in,
                              void* d_out, int out_size, void* d_ws,
                              size_t ws_size, hipStream_t stream) {
  const float* q  = (const float*)d_in[0];
  const float* k  = (const float*)d_in[1];
  const float* v  = (const float*)d_in[2];
  const float* xR = (const float*)d_in[3];
  const float* xT = (const float*)d_in[4];
  const float* zR = (const float*)d_in[5];
  const float* zT = (const float*)d_in[6];

  f16* kx = (f16*)d_ws;                      // 8 MB
  f16* vT = kx + (size_t)Bb * Hh * NK * CC;  // 8 MB
  f16* qx = vT + (size_t)Bb * Hh * NK * CC;  // 4 MB

  float* out  = (float*)d_out;
  float* attn = out + OUT_ELEMS;

  ipa_prep_kv<<<(Bb * Hh * NK * 16) / 256, 256, 0, stream>>>(k, v, zR, zT, kx,
                                                             vT);
  ipa_prep_q<<<(Bb * Hh * NQ * 16) / 256, 256, 0, stream>>>(q, xR, xT, qx);
  ipa_attn<<<Bb * Hh * (NQ / 128), 256, 0, stream>>>(qx, kx, vT, out, attn);
}
